// GotenNet_87316685128384
// MI455X (gfx1250) — compile-verified, matrix operand order checked by reading, not run on hardware
//
#include <hip/hip_runtime.h>

#define Nn 50000
#define Ee 800000

typedef __attribute__((ext_vector_type(16))) __bf16 v16bf;
typedef __attribute__((ext_vector_type(8)))  float  v8f;

union U16x16 { unsigned short u[16]; v16bf v; };

__device__ __forceinline__ unsigned short f2bf(float f) {
    union { __bf16 b; unsigned short u; } c; c.b = (__bf16)f; return c.u;
}
__device__ __forceinline__ float silu_f(float x) { return x / (1.0f + __expf(-x)); }

__device__ __forceinline__ v8f wmma_bf16(v16bf a, v16bf b, v8f c) {
    return __builtin_amdgcn_wmma_f32_16x16x32_bf16(false, a, false, b, (short)0, c, false, false);
}

// A fragment: 16 rows x 32 K from f32 global (row-clamped), vectorized b128 loads.
__device__ __forceinline__ v16bf ldA_g(const float* base, int row0, int maxrow, int ld, int k0) {
    int lane = threadIdx.x & 31;
    int row  = row0 + (lane & 15);
    row = row > maxrow ? maxrow : row;
    int kh = (lane & 16) ? 8 : 0;
    const float4* p = (const float4*)(base + (size_t)row * ld + k0 + kh);
    float4 x0 = p[0], x1 = p[1];   // K = k0+kh+0..7
    float4 y0 = p[4], y1 = p[5];   // K = k0+kh+16..23
    U16x16 r;
    r.u[0]  = f2bf(x0.x); r.u[1]  = f2bf(x0.y); r.u[2]  = f2bf(x0.z); r.u[3]  = f2bf(x0.w);
    r.u[4]  = f2bf(x1.x); r.u[5]  = f2bf(x1.y); r.u[6]  = f2bf(x1.z); r.u[7]  = f2bf(x1.w);
    r.u[8]  = f2bf(y0.x); r.u[9]  = f2bf(y0.y); r.u[10] = f2bf(y0.z); r.u[11] = f2bf(y0.w);
    r.u[12] = f2bf(y1.x); r.u[13] = f2bf(y1.y); r.u[14] = f2bf(y1.z); r.u[15] = f2bf(y1.w);
    return r.v;
}

// Fragment load from LDS in pre-swizzled per-lane layout: 16 contiguous bf16 per lane.
__device__ __forceinline__ v16bf ld_frag(const unsigned short* base, int frag) {
    int lane = threadIdx.x & 31;
    const uint4* p = (const uint4*)(base + (((size_t)frag * 32 + lane) << 4));
    U16x16 r;
    *(uint4*)(r.u)     = p[0];
    *(uint4*)(r.u + 8) = p[1];
    return r.v;
}

// Swizzle f32 weight [64 x ncols] into B-fragment layout in LDS:
// dst[((kt*nct + ct)*32 + lane)*16 + e], K = kt*32 + kh + (e&7) + (e&8 ? 16 : 0)
__device__ __forceinline__ void w2lds_frag(unsigned short* dst, const float* src,
                                           int ncols, int tid, int nth) {
    int nct = ncols >> 4;
    int total = 64 * ncols;
    for (int i = tid; i < total; i += nth) {
        int e    = i & 15;
        int lane = (i >> 4) & 31;
        int frag = i >> 9;
        int kt = frag / nct, ct = frag - kt * nct;
        int kh  = (lane & 16) ? 8 : 0;
        int col = lane & 15;
        int K = kt * 32 + kh + (e & 7) + ((e & 8) ? 16 : 0);
        dst[i] = f2bf(src[K * ncols + ct * 16 + col]);
    }
}

// ordered float atomic max via int/uint trick (works for all sign mixes)
__device__ __forceinline__ void atomicMaxF(float* addr, float v) {
    if (v >= 0.0f) atomicMax((int*)addr, __float_as_int(v));
    else           atomicMin((unsigned int*)addr, (unsigned int)__float_as_int(v));
}

// ---------------- K0: init ----------------
__global__ void k_init(const float* __restrict__ s, const float* __restrict__ t,
                       float* s_out, float* t_out, float* m, float* den) {
    int idx = blockIdx.x * blockDim.x + threadIdx.x;
    if (idx < Nn * 64)  s_out[idx] = s[idx];
    if (idx < Nn * 192) t_out[idx] = t[idx];
    if (idx < Nn * 8) { m[idx] = -3.0e38f; den[idx] = 0.0f; }
}

// ---------------- K1: q/k node GEMMs ----------------
__global__ void __launch_bounds__(128) k_qk(const float* __restrict__ s,
                                            const float* Wq, const float* __restrict__ bq,
                                            const float* Wk, const float* __restrict__ bk,
                                            float* q, float* kout) {
    __shared__ __align__(16) unsigned short wq[4096], wk[4096];
    int tid = threadIdx.x, nth = blockDim.x;
    w2lds_frag(wq, Wq, 64, tid, nth);
    w2lds_frag(wk, Wk, 64, tid, nth);
    __syncthreads();
    int wave = tid >> 5, lane = tid & 31, col = lane & 15, rb = (lane & 16) ? 8 : 0;
    int row0 = (blockIdx.x * 4 + wave) * 16;
    if (row0 >= Nn) return;
    v16bf a0 = ldA_g(s, row0, Nn - 1, 64, 0);
    v16bf a1 = ldA_g(s, row0, Nn - 1, 64, 32);
    bool full = (row0 + 16 <= Nn);
#pragma unroll
    for (int c = 0; c < 4; ++c) {
        v8f aq = {}; aq = wmma_bf16(a0, ld_frag(wq, c), aq);
        aq = wmma_bf16(a1, ld_frag(wq, 4 + c), aq);
        v8f ak = {}; ak = wmma_bf16(a0, ld_frag(wk, c), ak);
        ak = wmma_bf16(a1, ld_frag(wk, 4 + c), ak);
        float bqc = bq[c * 16 + col], bkc = bk[c * 16 + col];
        size_t o = (size_t)(row0 + rb) * 64 + c * 16 + col;
        if (full) {
#pragma unroll
            for (int r = 0; r < 8; ++r) {
                q[o + (size_t)r * 64]    = aq[r] + bqc;
                kout[o + (size_t)r * 64] = ak[r] + bkc;
            }
        } else {
#pragma unroll
            for (int r = 0; r < 8; ++r) {
                if (row0 + rb + r < Nn) {
                    q[o + (size_t)r * 64]    = aq[r] + bqc;
                    kout[o + (size_t)r * 64] = ak[r] + bkc;
                }
            }
        }
    }
}

// ---------------- K2: gated MLP chains (x and val) ----------------
__global__ void __launch_bounds__(128) k_gate(const float* __restrict__ s,
                                              const float* g1W, const float* __restrict__ g1b,
                                              const float* g2W, const float* __restrict__ g2b,
                                              const float* v1W, const float* __restrict__ v1b,
                                              const float* v2W, const float* __restrict__ v2b,
                                              float* xo, float* vo) {
    __shared__ __align__(16) unsigned short wIn[4096], wOut[12288];
    __shared__ __align__(16) unsigned short stage[4][1024];
    int tid = threadIdx.x, nth = blockDim.x;
    int wave = tid >> 5, lane = tid & 31, col = lane & 15, rb = (lane & 16) ? 8 : 0;
    int row0 = (blockIdx.x * 4 + wave) * 16;
    bool full = (row0 + 16 <= Nn);
    v16bf a0 = ldA_g(s, row0, Nn - 1, 64, 0);
    v16bf a1 = ldA_g(s, row0, Nn - 1, 64, 32);
    for (int pass = 0; pass < 2; ++pass) {
        const float* W1 = pass ? v1W : g1W; const float* B1 = pass ? v1b : g1b;
        const float* W2 = pass ? v2W : g2W; const float* B2 = pass ? v2b : g2b;
        float* out = pass ? vo : xo;
        __syncthreads();
        w2lds_frag(wIn, W1, 64, tid, nth);
        w2lds_frag(wOut, W2, 192, tid, nth);
        __syncthreads();
        unsigned short* st = stage[wave];
#pragma unroll
        for (int c = 0; c < 4; ++c) {
            v8f acc = {}; acc = wmma_bf16(a0, ld_frag(wIn, c), acc);
            acc = wmma_bf16(a1, ld_frag(wIn, 4 + c), acc);
            float bc = B1[c * 16 + col];
            // scatter silu(result) into A-fragment layout
            int kcol = c * 16 + col;
            int frag = kcol >> 5, krel = kcol & 31;
            int hi = (krel >> 3) & 1;
            int e  = (krel & 7) + ((krel & 16) ? 8 : 0);
            unsigned short* stp = st + (((frag * 32 + 16 * hi + rb) << 4) + e);
#pragma unroll
            for (int r = 0; r < 8; ++r)
                stp[r << 4] = f2bf(silu_f(acc[r] + bc));
        }
        v16bf h0 = ld_frag(st, 0);
        v16bf h1 = ld_frag(st, 1);
#pragma unroll
        for (int c = 0; c < 12; ++c) {
            v8f acc = {}; acc = wmma_bf16(h0, ld_frag(wOut, c), acc);
            acc = wmma_bf16(h1, ld_frag(wOut, 12 + c), acc);
            float bc = B2[c * 16 + col];
            size_t o = (size_t)(row0 + rb) * 192 + c * 16 + col;
            if (full) {
#pragma unroll
                for (int r = 0; r < 8; ++r) out[o + (size_t)r * 192] = acc[r] + bc;
            } else {
#pragma unroll
                for (int r = 0; r < 8; ++r)
                    if (row0 + rb + r < Nn) out[o + (size_t)r * 192] = acc[r] + bc;
            }
        }
    }
}

// ---------------- K4: fused phik GEMM + attention logits + seg-max ----------------
__global__ void __launch_bounds__(128) k_logits(const int* __restrict__ eidx,
                                                const float* __restrict__ rij,
                                                const float* phikW, const float* __restrict__ phikb,
                                                const float* __restrict__ q,
                                                const float* __restrict__ k,
                                                float* attn, float* m) {
    __shared__ __align__(16) unsigned short w[4096];
    __shared__ float ra[4][1024];
    int tid = threadIdx.x, nth = blockDim.x;
    w2lds_frag(w, phikW, 64, tid, nth);
    __syncthreads();
    int wave = tid >> 5, lane = tid & 31, col = lane & 15, rb = (lane & 16) ? 8 : 0;
    int e0 = (blockIdx.x * 4 + wave) * 16;
    v16bf a0 = ldA_g(rij, e0, Ee - 1, 64, 0);
    v16bf a1 = ldA_g(rij, e0, Ee - 1, 64, 32);
    float* ras = ra[wave];
#pragma unroll
    for (int c = 0; c < 4; ++c) {
        v8f acc = {}; acc = wmma_bf16(a0, ld_frag(w, c), acc);
        acc = wmma_bf16(a1, ld_frag(w, 4 + c), acc);
        float bc = phikb[c * 16 + col];
#pragma unroll
        for (int r = 0; r < 8; ++r)
            ras[(rb + r) * 64 + c * 16 + col] = silu_f(acc[r] + bc);
    }
    const int* jrow = eidx; const int* irow = eidx + Ee;
    for (int it = lane; it < 128; it += 32) {
        int el = it >> 3, h = it & 7;
        int e = e0 + el;
        if (e >= Ee) break;
        int i = irow[e], j = jrow[e];
        const float* qp = q + (size_t)i * 64 + h * 8;
        const float* kp = k + (size_t)j * 64 + h * 8;
        const float* rp = ras + el * 64 + h * 8;
        float dot = 0.0f;
#pragma unroll
        for (int d = 0; d < 8; ++d) dot += qp[d] * kp[d] * rp[d];
        attn[(size_t)e * 8 + h] = dot;
        atomicMaxF(&m[(size_t)i * 8 + h], dot);
    }
}

// ---------------- K5: exp + seg-sum ----------------
__global__ void k_expsum(const int* __restrict__ eidx, float* attn,
                         const float* __restrict__ m, float* den) {
    int t = blockIdx.x * blockDim.x + threadIdx.x;
    if (t >= Ee * 8) return;
    int e = t >> 3, h = t & 7;
    int i = eidx[Ee + e];
    float w = __expf(attn[t] - m[(size_t)i * 8 + h]);
    attn[t] = w;
    atomicAdd(&den[(size_t)i * 8 + h], w);
}

// ---------------- K6: fused wre GEMM + messages + scatter ----------------
__global__ void __launch_bounds__(128) k_msg(const int* __restrict__ eidx,
                                             const float* __restrict__ rij,
                                             const float* __restrict__ dij,
                                             const float* __restrict__ deg,
                                             const float* wreW, const float* __restrict__ wreb,
                                             const float* __restrict__ xw,
                                             const float* __restrict__ vw,
                                             const float* __restrict__ tin,
                                             const float* __restrict__ dir,
                                             const float* __restrict__ attn,
                                             const float* __restrict__ den,
                                             float* s_out, float* t_out) {
    __shared__ __align__(16) unsigned short w[12288];
    __shared__ float rl[4][3072];
    __shared__ float coef[4][128];
    __shared__ float cutw[4][16];
    int tid = threadIdx.x, nth = blockDim.x;
    w2lds_frag(w, wreW, 192, tid, nth);
    __syncthreads();
    int wave = tid >> 5, lane = tid & 31, col = lane & 15, rb = (lane & 16) ? 8 : 0;
    int e0 = (blockIdx.x * 4 + wave) * 16;
    const int* jrow = eidx; const int* irow = eidx + Ee;
    if (lane < 16) {
        int e = e0 + lane; float cut = 0.0f;
        if (e < Ee) { float d = dij[e]; cut = (d < 5.0f) ? 0.5f * (__cosf(0.62831853072f * d) + 1.0f) : 0.0f; }
        cutw[wave][lane] = cut;
    }
    for (int it = lane; it < 128; it += 32) {
        int el = it >> 3, h = it & 7; int e = e0 + el;
        float cf = 0.0f;
        if (e < Ee) {
            int i = irow[e];
            cf = attn[(size_t)e * 8 + h] / (den[(size_t)i * 8 + h] + 1e-16f) * sqrtf(deg[e]) * 0.125f;
        }
        coef[wave][it] = cf;
    }
    v16bf a0 = ldA_g(rij, e0, Ee - 1, 64, 0);
    v16bf a1 = ldA_g(rij, e0, Ee - 1, 64, 32);
    float* rls = rl[wave];
#pragma unroll
    for (int c = 0; c < 12; ++c) {
        v8f acc = {}; acc = wmma_bf16(a0, ld_frag(w, c), acc);
        acc = wmma_bf16(a1, ld_frag(w, 12 + c), acc);
        float bc = wreb[c * 16 + col];
#pragma unroll
        for (int r = 0; r < 8; ++r) {
            int el = rb + r;
            rls[el * 192 + c * 16 + col] = (acc[r] + bc) * cutw[wave][el];
        }
    }
    for (int it = lane; it < 1024; it += 32) {
        int el = it >> 6, f = it & 63;
        int e = e0 + el;
        if (e >= Ee) break;
        int i = irow[e], j = jrow[e];
        const float* vj = vw + (size_t)j * 192;
        const float* xj = xw + (size_t)j * 192;
        const float* tj = tin + (size_t)j * 192;
        const float* cf = coef[wave] + el * 8;
        const float* rle = rls + el * 192;
        float xm0 = cf[f / 24]         * vj[f]       + rle[f]       * xj[f];
        float xm1 = cf[(64 + f) / 24]  * vj[64 + f]  + rle[64 + f]  * xj[64 + f];
        float xm2 = cf[(128 + f) / 24] * vj[128 + f] + rle[128 + f] * xj[128 + f];
        atomicAdd(&s_out[(size_t)i * 64 + f], xm0);
        float d0 = dir[e * 3 + 0], d1 = dir[e * 3 + 1], d2 = dir[e * 3 + 2];
        atomicAdd(&t_out[(size_t)i * 192 +       f], xm1 * d0 + xm2 * tj[f]);
        atomicAdd(&t_out[(size_t)i * 192 +  64 + f], xm1 * d1 + xm2 * tj[64 + f]);
        atomicAdd(&t_out[(size_t)i * 192 + 128 + f], xm1 * d2 + xm2 * tj[128 + f]);
    }
}

// ---------------- K7: vecq/veck GEMMs over t_out (3N rows) ----------------
__global__ void __launch_bounds__(128) k_vec(const float* __restrict__ tarr,
                                             const float* Wa, const float* Wb,
                                             float* w1, float* w2) {
    __shared__ __align__(16) unsigned short wa[4096], wb[4096];
    int tid = threadIdx.x, nth = blockDim.x;
    w2lds_frag(wa, Wa, 64, tid, nth);
    w2lds_frag(wb, Wb, 64, tid, nth);
    __syncthreads();
    const int R = 3 * Nn;
    int wave = tid >> 5, lane = tid & 31, col = lane & 15, rb = (lane & 16) ? 8 : 0;
    int row0 = (blockIdx.x * 4 + wave) * 16;
    if (row0 >= R) return;
    v16bf a0 = ldA_g(tarr, row0, R - 1, 64, 0);
    v16bf a1 = ldA_g(tarr, row0, R - 1, 64, 32);
    bool full = (row0 + 16 <= R);
#pragma unroll
    for (int c = 0; c < 4; ++c) {
        v8f s1 = {}; s1 = wmma_bf16(a0, ld_frag(wa, c), s1);
        s1 = wmma_bf16(a1, ld_frag(wa, 4 + c), s1);
        v8f s2 = {}; s2 = wmma_bf16(a0, ld_frag(wb, c), s2);
        s2 = wmma_bf16(a1, ld_frag(wb, 4 + c), s2);
        size_t o = (size_t)(row0 + rb) * 64 + c * 16 + col;
        if (full) {
#pragma unroll
            for (int r = 0; r < 8; ++r) {
                w1[o + (size_t)r * 64] = s1[r];
                w2[o + (size_t)r * 64] = s2[r];
            }
        } else {
#pragma unroll
            for (int r = 0; r < 8; ++r) {
                if (row0 + rb + r < R) {
                    w1[o + (size_t)r * 64] = s1[r];
                    w2[o + (size_t)r * 64] = s2[r];
                }
            }
        }
    }
}

// ---------------- K8: fused eup GEMM + rejection dot + df_ij ----------------
__global__ void __launch_bounds__(128) k_edge(const int* __restrict__ eidx,
                                              const float* __restrict__ rij,
                                              const float* eupW, const float* __restrict__ eupb,
                                              const float* __restrict__ w1,
                                              const float* __restrict__ w2,
                                              const float* __restrict__ dir,
                                              float* dfo) {
    __shared__ __align__(16) unsigned short w[4096];
    __shared__ float gw[4][1024];
    int tid = threadIdx.x, nth = blockDim.x;
    w2lds_frag(w, eupW, 64, tid, nth);
    __syncthreads();
    int wave = tid >> 5, lane = tid & 31, col = lane & 15, rb = (lane & 16) ? 8 : 0;
    int e0 = (blockIdx.x * 4 + wave) * 16;
    v16bf a0 = ldA_g(rij, e0, Ee - 1, 64, 0);
    v16bf a1 = ldA_g(rij, e0, Ee - 1, 64, 32);
    float* gws = gw[wave];
#pragma unroll
    for (int c = 0; c < 4; ++c) {
        v8f acc = {}; acc = wmma_bf16(a0, ld_frag(w, c), acc);
        acc = wmma_bf16(a1, ld_frag(w, 4 + c), acc);
        float bc = eupb[c * 16 + col];
#pragma unroll
        for (int r = 0; r < 8; ++r)
            gws[(rb + r) * 64 + c * 16 + col] = silu_f(acc[r] + bc);
    }
    const int* jrow = eidx; const int* irow = eidx + Ee;
    for (int it = lane; it < 1024; it += 32) {
        int el = it >> 6, f = it & 63;
        int e = e0 + el;
        if (e >= Ee) break;
        int i = irow[e], j = jrow[e];
        float d0 = dir[e * 3 + 0], d1 = dir[e * 3 + 1], d2 = dir[e * 3 + 2];
        const float* ap = w1 + (size_t)i * 192;
        const float* bp = w2 + (size_t)j * 192;
        float a0v = ap[f], a1v = ap[64 + f], a2v = ap[128 + f];
        float b0v = bp[f], b1v = bp[64 + f], b2v = bp[128 + f];
        float p1 = a0v * d0 + a1v * d1 + a2v * d2;
        float p2 = b0v * d0 + b1v * d1 + b2v * d2;
        float wd = (a0v - p1 * d0) * (b0v - p2 * d0)
                 + (a1v - p1 * d1) * (b1v - p2 * d1)
                 + (a2v - p1 * d2) * (b2v - p2 * d2);
        dfo[(size_t)e * 64 + f] = rij[(size_t)e * 64 + f] + gws[el * 64 + f] * wd;
    }
}

extern "C" void kernel_launch(void* const* d_in, const int* in_sizes, int n_in,
                              void* d_out, int out_size, void* d_ws, size_t ws_size,
                              hipStream_t stream) {
    (void)in_sizes; (void)n_in; (void)out_size; (void)ws_size;
    const int*   eidx = (const int*)d_in[0];
    const float* s    = (const float*)d_in[1];
    const float* t    = (const float*)d_in[2];
    const float* dir  = (const float*)d_in[3];
    const float* rij  = (const float*)d_in[4];
    const float* dij  = (const float*)d_in[5];
    const float* deg  = (const float*)d_in[6];
    const float* Wq   = (const float*)d_in[7];  const float* bq   = (const float*)d_in[8];
    const float* Wk   = (const float*)d_in[9];  const float* bk   = (const float*)d_in[10];
    const float* gs1W = (const float*)d_in[11]; const float* gs1b = (const float*)d_in[12];
    const float* gs2W = (const float*)d_in[13]; const float* gs2b = (const float*)d_in[14];
    const float* gv1W = (const float*)d_in[15]; const float* gv1b = (const float*)d_in[16];
    const float* gv2W = (const float*)d_in[17]; const float* gv2b = (const float*)d_in[18];
    const float* phikW= (const float*)d_in[19]; const float* phikb= (const float*)d_in[20];
    const float* wreW = (const float*)d_in[21]; const float* wreb = (const float*)d_in[22];
    const float* eupW = (const float*)d_in[23]; const float* eupb = (const float*)d_in[24];
    const float* vecqW= (const float*)d_in[25]; const float* veckW= (const float*)d_in[26];

    float* ws   = (float*)d_ws;
    float* q    = ws + 0;
    float* kk   = ws + 3200000;   // N*64
    float* x    = ws + 6400000;   // N*192
    float* val  = ws + 16000000;  // N*192
    float* attn = ws + 25600000;  // E*8
    float* m    = ws + 32000000;  // N*8
    float* den  = ws + 32400000;  // N*8
    float* w1   = ws + 32800000;  // N*192
    float* w2   = ws + 42400000;  // N*192

    float* s_out = (float*)d_out;
    float* t_out = s_out + (size_t)Nn * 64;
    float* dfo   = t_out + (size_t)Nn * 192;

    k_init  <<<(Nn * 192 + 255) / 256, 256, 0, stream>>>(s, t, s_out, t_out, m, den);
    k_qk    <<<(Nn + 63) / 64, 128, 0, stream>>>(s, Wq, bq, Wk, bk, q, kk);
    k_gate  <<<(Nn + 63) / 64, 128, 0, stream>>>(s, gs1W, gs1b, gs2W, gs2b,
                                                 gv1W, gv1b, gv2W, gv2b, x, val);
    k_logits<<<(Ee + 63) / 64, 128, 0, stream>>>(eidx, rij, phikW, phikb, q, kk, attn, m);
    k_expsum<<<(Ee * 8 + 255) / 256, 256, 0, stream>>>(eidx, attn, m, den);
    k_msg   <<<(Ee + 63) / 64, 128, 0, stream>>>(eidx, rij, dij, deg, wreW, wreb,
                                                 x, val, t, dir, attn, den, s_out, t_out);
    k_vec   <<<(3 * Nn + 63) / 64, 128, 0, stream>>>(t_out, vecqW, veckW, w1, w2);
    k_edge  <<<(Ee + 63) / 64, 128, 0, stream>>>(eidx, rij, eupW, eupb, w1, w2, dir, dfo);
}
